// KP_Decoder_28647431864766
// MI455X (gfx1250) — matrix operand order repaired and checked by reading, not compiled, not run on hardware
//
#include <hip/hip_runtime.h>
#include <hip/hip_bf16.h>

// ---------------------------------------------------------------------------
// Problem constants (from reference)
// ---------------------------------------------------------------------------
#define Bb   64
#define Pp   200
#define Nn   2000
#define Dd   128
#define NPAD 2048          // N padded to multiple of 32 for the WMMA K loop
#define PTIL 13            // ceil(P/16)
#define ISQRT_D 0.08838834764831845f   // 1/sqrt(128)
#define CLIPV 10.0f

// ---------------------------------------------------------------------------
// WMMA types / helpers (gfx1250, wave32)
// ---------------------------------------------------------------------------
typedef __bf16 v16bf __attribute__((ext_vector_type(16)));
typedef float  v8f   __attribute__((ext_vector_type(8)));
typedef unsigned int u32x4 __attribute__((ext_vector_type(4)));
typedef int          i32x4 __attribute__((ext_vector_type(4)));
typedef int          i32x8 __attribute__((ext_vector_type(8)));

union BFPack {
    v16bf v;
    unsigned short s[16];
    uint4 u[2];
};

__device__ inline unsigned short bfbits(float x) {
    union { __bf16 h; unsigned short u; } t;
    t.h = (__bf16)x;            // native f32->bf16 convert
    return t.u;
}

__device__ inline v8f wmma_bf16(v16bf a, v16bf b, v8f c) {
    // D = A(16x32 bf16) x B(32x16 bf16) + C(16x16 f32)
    return __builtin_amdgcn_wmma_f32_16x16x32_bf16(
        /*neg_a=*/false, a, /*neg_b=*/false, b,
        /*c_mod=*/(short)0, c, /*reuse_a=*/false, /*reuse_b=*/false);
}

__device__ inline v16bf cvt16(float4 a, float4 b, float4 c, float4 d) {
    v16bf r;
    r[0]  = (__bf16)a.x; r[1]  = (__bf16)a.y; r[2]  = (__bf16)a.z; r[3]  = (__bf16)a.w;
    r[4]  = (__bf16)b.x; r[5]  = (__bf16)b.y; r[6]  = (__bf16)b.z; r[7]  = (__bf16)b.w;
    r[8]  = (__bf16)c.x; r[9]  = (__bf16)c.y; r[10] = (__bf16)c.z; r[11] = (__bf16)c.w;
    r[12] = (__bf16)d.x; r[13] = (__bf16)d.y; r[14] = (__bf16)d.z; r[15] = (__bf16)d.w;
    return r;
}

// A fragment from a row-major f32 row: elems 0..15 = K k0..k0+15,
// with k0 = kc + (hi?16:0)  (contiguous K order, matches B fragments).
__device__ inline v16bf ldA_f32(const float* rowp, int kc, int hi) {
    int k0 = kc + (hi ? 16 : 0);
    const float4* q0 = (const float4*)(rowp + k0);
    return cvt16(q0[0], q0[1], q0[2], q0[3]);
}

// B fragment from 16 contiguous f32 (lane holds K = kbase+(hi?16:0) .. +15, col = lane%16)
__device__ inline v16bf ldB_f32(const float* p) {
    const float4* q = (const float4*)p;
    return cvt16(q[0], q[1], q[2], q[3]);
}

// 16 contiguous pre-packed bf16 (staged fragments)
__device__ inline v16bf ldB_bf16raw(const void* p) {
    BFPack r;
    r.u[0] = *(const uint4*)(p);
    r.u[1] = *(const uint4*)((const char*)p + 16);
    return r.v;
}

// A fragment from a row-major bf16 row in LDS
__device__ inline v16bf ldA_bf16(const unsigned short* rowp, int kc, int hi) {
    int k0 = kc + (hi ? 16 : 0);
    BFPack r;
    r.u[0] = *(const uint4*)(rowp + k0);
    r.u[1] = *(const uint4*)(rowp + k0 + 8);
    return r.v;
}

// ---------------------------------------------------------------------------
// gfx1250 async copy: 16B per lane, global -> LDS, tracked by ASYNCcnt
// ---------------------------------------------------------------------------
__device__ inline void async_cp32(const unsigned short* g, unsigned lds) {
    asm volatile("global_load_async_to_lds_b128 %0, %1, off"
                 :: "v"(lds), "v"(g) : "memory");
    asm volatile("global_load_async_to_lds_b128 %0, %1, off"
                 :: "v"(lds + 16u), "v"(g + 8) : "memory");
}

// ---------------------------------------------------------------------------
// Kernel 1: ek = exp(enc·Wk^T), ekv = ek * (enc·Wv^T), stored TRANSPOSED bf16
// ekT[b][e][n], ekvT[b][e][n], n padded to NPAD (pad region zeroed).
// Wk/Wv are staged into LDS by the Tensor Data Mover (TDM), then consumed as
// WMMA B fragments.  Block = 128 thr (4 waves): one 16-row n tile per block,
// wave w covers e-range [32w, 32w+32).
// ---------------------------------------------------------------------------
__device__ inline void tdm_load_128x128_f32(const float* gsrc, unsigned lds_byte_addr) {
    // D# group 0: count=1, lds_addr, 57-bit global addr, type=2 ("image")
    unsigned long long ga = (unsigned long long)(const void*)gsrc;
    u32x4 g0;
    g0[0] = 1u;                                   // count=1, is_restore=0
    g0[1] = lds_byte_addr;                        // lds_addr [63:32]
    g0[2] = (unsigned)(ga & 0xFFFFFFFFu);         // global_addr [95:64]
    g0[3] = (unsigned)((ga >> 32) & 0x1FFFFFFu)   // global_addr [120:96]
          | (2u << 30);                           // type=2 [127:126]
    // D# group 1: data_size=4B, tensor 128x128, tile 128x128, stride0=128
    const unsigned dim0 = 128, dim1 = 128, tile0 = 128, tile1 = 128, stride0 = 128;
    i32x8 g1;
    g1[0] = (int)(2u << 16);                      // workgroup_mask=0, data_size=2 (4B)
    g1[1] = (int)((dim0 & 0xFFFFu) << 16);        // tensor_dim0[15:0] @ bits 63:48
    g1[2] = (int)(((dim0 >> 16) & 0xFFFFu) | ((dim1 & 0xFFFFu) << 16));
    g1[3] = (int)(((dim1 >> 16) & 0xFFFFu) | (tile0 << 16));   // tile_dim0 @ 127:112
    g1[4] = (int)(tile1);                         // tile_dim1 @ 143:128, tile_dim2=0
    g1[5] = (int)(stride0);                       // tensor_dim0_stride[31:0]
    g1[6] = 0;                                    // stride0 hi / dim1_stride lo
    g1[7] = 0;
    i32x4 z4 = {0, 0, 0, 0};
    i32x8 z8 = {0, 0, 0, 0, 0, 0, 0, 0};
    // 6-arg form (clang-23 / therock-10.0 headers): groups 0..3 + extra + cpol
    __builtin_amdgcn_tensor_load_to_lds(g0, g1, z4, z4, z8, 0);
}

__global__ __launch_bounds__(128) void setkv_kernel(
    const float* __restrict__ enc, const float* __restrict__ Wk,
    const float* __restrict__ Wv,
    unsigned short* __restrict__ ekT, unsigned short* __restrict__ ekvT)
{
    __shared__ __align__(16) float wkbuf[Dd * Dd];   // 64 KB
    __shared__ __align__(16) float wvbuf[Dd * Dd];   // 64 KB

    int blk  = blockIdx.x;
    int b    = blk >> 7;          // 128 n-tiles per batch
    int nb   = (blk & 127) * 16;
    int wave = threadIdx.x >> 5;
    int lane = threadIdx.x & 31;
    int col  = lane & 15;
    int hi   = lane >> 4;
    int ebase = wave * 32;

    if (nb >= Nn) {  // zero the pad region (n in [2000,2048))
        uint4 z{0, 0, 0, 0};
        for (int t = 0; t < 2; ++t) {
            int e = ebase + t * 16 + col;
            size_t off = ((size_t)b * Dd + e) * NPAD + nb + (hi ? 8 : 0);
            *(uint4*)(ekT + off)  = z;
            *(uint4*)(ekvT + off) = z;
        }
        return;
    }

    // --- TDM: pull Wk and Wv into LDS (one wave issues; TENSORcnt tracked) ---
    if (wave == 0) {
        tdm_load_128x128_f32(Wk, (unsigned)(uintptr_t)(void*)wkbuf);
        tdm_load_128x128_f32(Wv, (unsigned)(uintptr_t)(void*)wvbuf);
        __builtin_amdgcn_s_wait_tensorcnt(0);
    }
    __syncthreads();

    v8f k0{}, k1{}, v0{}, v1{};
    const float* arow  = enc + ((size_t)b * Nn + nb + col) * Dd;  // A row: n = nb+col
    const float* wkrow = wkbuf + (size_t)(ebase + col) * Dd + (hi ? 16 : 0);
    const float* wvrow = wvbuf + (size_t)(ebase + col) * Dd + (hi ? 16 : 0);

    #pragma unroll
    for (int kc = 0; kc < Dd; kc += 32) {
        v16bf a = ldA_f32(arow, kc, hi);
        k0 = wmma_bf16(a, ldB_f32(wkrow + kc),            k0);
        k1 = wmma_bf16(a, ldB_f32(wkrow + 16 * Dd + kc),  k1);
        v0 = wmma_bf16(a, ldB_f32(wvrow + kc),            v0);
        v1 = wmma_bf16(a, ldB_f32(wvrow + 16 * Dd + kc),  v1);
    }

    // C layout: VGPR r -> n_local = r + (hi?8:0), col -> e.  8 consecutive n per lane
    // pack into one 16B store into the transposed [e][n] layout.
    #pragma unroll
    for (int t = 0; t < 2; ++t) {
        v8f kk = t ? k1 : k0;
        v8f vv = t ? v1 : v0;
        BFPack pk, pkv;
        #pragma unroll
        for (int r = 0; r < 8; ++r) {
            float ek = __expf(kk[r]);
            pk.s[r]  = bfbits(ek);
            pkv.s[r] = bfbits(ek * vv[r]);
        }
        int e = ebase + t * 16 + col;
        size_t off = ((size_t)b * Dd + e) * NPAD + nb + (hi ? 8 : 0);
        *(uint4*)(ekT + off)  = pk.u[0];
        *(uint4*)(ekvT + off) = pkv.u[0];
    }
}

// ---------------------------------------------------------------------------
// Kernel 2: sigmoid_q[b,p,e] = sigmoid( [enc_mean|capacity] · Wq_last^T )
// One block per (b,p) row, 128 threads = one output element each.
// ---------------------------------------------------------------------------
__global__ __launch_bounds__(128) void query_kernel(
    const float* __restrict__ mean, const float* __restrict__ cap,
    const float* __restrict__ Wq, float* __restrict__ sq)
{
    __shared__ float row[Dd + 1];
    int bp = blockIdx.x;            // b*P + p
    int e  = threadIdx.x;
    row[e] = mean[(size_t)bp * Dd + e];
    if (e == 0) row[Dd] = cap[bp];
    __syncthreads();
    const float* w = Wq + (size_t)e * (Dd + 1);
    float s = 0.f;
    #pragma unroll 4
    for (int f = 0; f < Dd + 1; ++f) s += row[f] * w[f];
    sq[(size_t)bp * Dd + e] = 1.f / (1.f + __expf(-s));
}

// ---------------------------------------------------------------------------
// Kernel 3: fused AFT weighting + score GEMM + tanh clip + softmax.
// One block (4 waves) per (b, 16-row p tile).
// ---------------------------------------------------------------------------
__device__ inline v16bf ea_fast(const float* cd, const float* nm, int n0, float s) {
    const float4* c = (const float4*)(cd + n0);
    const float4* m = (const float4*)(nm + n0);
    float4 c0 = c[0], c1 = c[1], c2 = c[2], c3 = c[3];
    float4 m0 = m[0], m1 = m[1], m2 = m[2], m3 = m[3];
    v16bf r;
    r[0]  = (__bf16)__expf(s * c0.x + m0.x); r[1]  = (__bf16)__expf(s * c0.y + m0.y);
    r[2]  = (__bf16)__expf(s * c0.z + m0.z); r[3]  = (__bf16)__expf(s * c0.w + m0.w);
    r[4]  = (__bf16)__expf(s * c1.x + m1.x); r[5]  = (__bf16)__expf(s * c1.y + m1.y);
    r[6]  = (__bf16)__expf(s * c1.z + m1.z); r[7]  = (__bf16)__expf(s * c1.w + m1.w);
    r[8]  = (__bf16)__expf(s * c2.x + m2.x); r[9]  = (__bf16)__expf(s * c2.y + m2.y);
    r[10] = (__bf16)__expf(s * c2.z + m2.z); r[11] = (__bf16)__expf(s * c2.w + m2.w);
    r[12] = (__bf16)__expf(s * c3.x + m3.x); r[13] = (__bf16)__expf(s * c3.y + m3.y);
    r[14] = (__bf16)__expf(s * c3.z + m3.z); r[15] = (__bf16)__expf(s * c3.w + m3.w);
    return r;
}

__device__ inline v16bf ea_guard(const float* cd, const float* nm, int n0, float s) {
    v16bf r;
    #pragma unroll
    for (int i = 0; i < 16; ++i) {
        int n = n0 + i;
        float e = 0.f;
        if (n < Nn) e = __expf(s * cd[n] + nm[n]);
        r[i] = (__bf16)e;
    }
    return r;
}

__global__ __launch_bounds__(128) void aft_kernel(
    const float* __restrict__ cur_dist, const float* __restrict__ ninf,
    const float* __restrict__ enc,
    const unsigned short* __restrict__ ekT, const unsigned short* __restrict__ ekvT,
    const float* __restrict__ sq,
    const float* __restrict__ ls_p, const float* __restrict__ a1_p,
    const float* __restrict__ a2_p, float* __restrict__ out)
{
    __shared__ __align__(16) float score[16 * Nn];        // 128 KB (CDNA5: 320 KB/WGP)
    __shared__ __align__(16) unsigned short aft[16 * Dd]; // 4 KB bf16 aft tile

    int blk   = blockIdx.x;
    int b     = blk / PTIL;
    int pbase = (blk % PTIL) * 16;
    int tid   = threadIdx.x;
    int wave  = tid >> 5;
    int lane  = tid & 31;
    int col   = lane & 15;
    int hi    = lane >> 4;

    float ls     = ls_p[0];
    float scale1 = ls * a1_p[0];
    float scale2 = ls * a2_p[0];

    // ----- Phase 1: bias/denom = ea · ekv , ea · ek  (K loop over N) -----
    int prow   = pbase + col;
    int pclamp = prow < Pp ? prow : Pp - 1;    // tail tile rows duplicate p=199 (unused)
    const float* cd_row = cur_dist + ((size_t)b * Pp + pclamp) * Nn;
    const float* nm_row = ninf     + ((size_t)b * Pp + pclamp) * Nn;

    int dbase = wave * 32;                      // each wave owns 32 of the 128 d columns
    const unsigned short* ekb  = ekT  + ((size_t)b * Dd + dbase + col) * NPAD + (hi ? 16 : 0);
    const unsigned short* ekvb = ekvT + ((size_t)b * Dd + dbase + col) * NPAD + (hi ? 16 : 0);

    // Async double-buffered staging for the 4 B-fragment streams.
    // Reuses the (phase-2) score LDS: wave w owns bytes [w*8K, w*8K+8K).
    // Buffer layout: buf (4KB) = 4 fragments x (32 lanes x 32B).
    char* stage = (char*)score + wave * 8192 + lane * 32;
    unsigned stage_lds0 = (unsigned)(uintptr_t)(void*)stage;           // buffer 0
    unsigned stage_lds1 = stage_lds0 + 4096;                           // buffer 1

    // prologue: stage step 0 into buffer 0
    {
        async_cp32(ekb,               stage_lds0);
        async_cp32(ekb  + 16 * NPAD,  stage_lds0 + 1024);
        async_cp32(ekvb,              stage_lds0 + 2048);
        async_cp32(ekvb + 16 * NPAD,  stage_lds0 + 3072);
    }

    v8f accB0{}, accB1{}, accD0{}, accD1{};
    for (int i = 0; i < NPAD / 32; ++i) {
        int nb = i * 32;
        if (i < NPAD / 32 - 1) {            // stage next step into the other buffer
            int nb2 = nb + 32;
            unsigned dst = (i & 1) ? stage_lds0 : stage_lds1;
            async_cp32(ekb  + nb2,             dst);
            async_cp32(ekb  + 16 * NPAD + nb2, dst + 1024);
            async_cp32(ekvb + nb2,             dst + 2048);
            async_cp32(ekvb + 16 * NPAD + nb2, dst + 3072);
            asm volatile("s_wait_asynccnt 0x8" ::: "memory");
        } else {
            asm volatile("s_wait_asynccnt 0x0" ::: "memory");
        }

        if (nb + 288 < Nn) {               // stream hints for the next A fragments
            __builtin_prefetch(cd_row + nb + 256, 0, 1);
            __builtin_prefetch(nm_row + nb + 256, 0, 1);
        }
        v16bf a = (nb < 1984) ? ea_fast(cd_row, nm_row, nb + (hi ? 16 : 0), scale1)
                              : ea_guard(cd_row, nm_row, nb + (hi ? 16 : 0), scale1);

        const char* buf = (char*)score + wave * 8192 + ((i & 1) ? 4096 : 0) + lane * 32;
        accD0 = wmma_bf16(a, ldB_bf16raw(buf),        accD0);
        accD1 = wmma_bf16(a, ldB_bf16raw(buf + 1024), accD1);
        accB0 = wmma_bf16(a, ldB_bf16raw(buf + 2048), accB0);
        accB1 = wmma_bf16(a, ldB_bf16raw(buf + 3072), accB1);
    }

    // weighted = bias/denom (with nan_to_num fallback), aft = sigmoid_q * weighted -> LDS
    const float* sqb = sq + (size_t)b * Pp * Dd;
    #pragma unroll
    for (int t = 0; t < 2; ++t) {
        v8f bb = t ? accB1 : accB0;
        v8f dd = t ? accD1 : accD0;
        int e = dbase + t * 16 + col;
        #pragma unroll
        for (int r = 0; r < 8; ++r) {
            int pl = r + (hi ? 8 : 0);
            int pg = pbase + pl;
            int pc = pg < Pp ? pg : Pp - 1;
            float bias = bb[r], den = dd[r];
            float w = bias / den;
            if (w != w) {
                float b2 = (bias != bias) ? 0.f : bias;
                float d2 = (den  != den)  ? 0.f : den;
                w = b2 / d2;
            }
            float s = sqb[(size_t)pc * Dd + e];
            aft[pl * Dd + e] = bfbits(s * w);
        }
    }
    __syncthreads();   // staging region dead from here; score LDS free for phase 2

    // ----- Phase 2: score = aft · enc^T / sqrt(D) + dist bias, clipped tanh -----
    const unsigned short* arow = aft + col * Dd;   // A row: p = lane%16
    v16bf A0 = ldA_bf16(arow, 0,  hi);
    v16bf A1 = ldA_bf16(arow, 32, hi);
    v16bf A2 = ldA_bf16(arow, 64, hi);
    v16bf A3 = ldA_bf16(arow, 96, hi);

    for (int t = wave; t < Nn / 16; t += 4) {      // 125 n-tiles round-robin over waves
        int nb2 = t * 16;
        const float* brow = enc + ((size_t)b * Nn + nb2 + col) * Dd + (hi ? 16 : 0);
        v8f acc{};
        acc = wmma_bf16(A0, ldB_f32(brow),      acc);
        acc = wmma_bf16(A1, ldB_f32(brow + 32), acc);
        acc = wmma_bf16(A2, ldB_f32(brow + 64), acc);
        acc = wmma_bf16(A3, ldB_f32(brow + 96), acc);
        int n = nb2 + col;
        #pragma unroll
        for (int r = 0; r < 8; ++r) {
            int pl = r + (hi ? 8 : 0);
            int pg = pbase + pl;
            int pc = pg < Pp ? pg : Pp - 1;
            size_t ridx = ((size_t)b * Pp + pc) * Nn + n;
            float s = acc[r] * ISQRT_D + scale2 * cur_dist[ridx];
            s = CLIPV * tanhf(s) + ninf[ridx];
            score[pl * Nn + n] = s;
        }
    }
    __syncthreads();

    // ----- Softmax over N per row (8 threads/row, shfl_xor group reductions) -----
    int row = tid >> 3, sub = tid & 7;
    float* srow = score + row * Nn;
    float mx = -3.4e38f;
    for (int n = sub; n < Nn; n += 8) mx = fmaxf(mx, srow[n]);
    #pragma unroll
    for (int o = 1; o < 8; o <<= 1) mx = fmaxf(mx, __shfl_xor(mx, o, 8));
    float sum = 0.f;
    for (int n = sub; n < Nn; n += 8) {
        float e = __expf(srow[n] - mx);
        srow[n] = e;
        sum += e;
    }
    #pragma unroll
    for (int o = 1; o < 8; o <<= 1) sum += __shfl_xor(sum, o, 8);
    float inv = 1.f / sum;
    int pg = pbase + row;
    if (pg < Pp) {
        float* orow = out + ((size_t)b * Pp + pg) * Nn;
        for (int n = sub; n < Nn; n += 8) orow[n] = srow[n] * inv;
    }
}

// ---------------------------------------------------------------------------
// Launcher. Workspace layout: ekT (33.5 MB bf16) | ekvT (33.5 MB) | sq (6.5 MB f32)
// ---------------------------------------------------------------------------
extern "C" void kernel_launch(void* const* d_in, const int* in_sizes, int n_in,
                              void* d_out, int out_size, void* d_ws, size_t ws_size,
                              hipStream_t stream) {
    const float* mean = (const float*)d_in[0];   // encoded_graph_mean_pomo [B,P,D]
    const float* cap  = (const float*)d_in[1];   // capacity [B,P]
    const float* cur  = (const float*)d_in[2];   // cur_dist [B,P,N]
    const float* ls   = (const float*)d_in[3];   // log_scale [1]
    const float* nm   = (const float*)d_in[4];   // ninf_mask [B,P,N]
    const float* enc  = (const float*)d_in[5];   // encoded_nodes [B,N,D]
    const float* Wq   = (const float*)d_in[6];   // [D, D+1]
    const float* Wk   = (const float*)d_in[7];   // [D, D]
    const float* Wv   = (const float*)d_in[8];   // [D, D]
    const float* a1   = (const float*)d_in[9];   // AFT_dist_alpha [1]
    const float* a2   = (const float*)d_in[10];  // probs_dist_alpha [1]
    float* out = (float*)d_out;

    char* ws = (char*)d_ws;
    unsigned short* ekT  = (unsigned short*)ws;
    unsigned short* ekvT = (unsigned short*)(ws + (size_t)Bb * Dd * NPAD * 2);
    float*          sq   = (float*)(ws + (size_t)Bb * Dd * NPAD * 4);

    hipLaunchKernelGGL(setkv_kernel, dim3(Bb * (NPAD / 16)), dim3(128), 0, stream,
                       enc, Wk, Wv, ekT, ekvT);
    hipLaunchKernelGGL(query_kernel, dim3(Bb * Pp), dim3(128), 0, stream,
                       mean, cap, Wq, sq);
    hipLaunchKernelGGL(aft_kernel, dim3(Bb * PTIL), dim3(128), 0, stream,
                       cur, nm, enc, ekT, ekvT, sq, ls, a1, a2, out);
}
